// MNISTMonDEQ_4037269258580
// MI455X (gfx1250) — matrix operand order repaired and checked by reading, not compiled, not run on hardware
//
#include <hip/hip_runtime.h>
#include <hip/hip_bf16.h>

typedef __attribute__((ext_vector_type(16))) _Float16 v16h;
typedef __attribute__((ext_vector_type(8)))  _Float16 v8h;
typedef __attribute__((ext_vector_type(8)))  float    v8f;

#define N_DIM 512
#define D_DIM 784
#define BATCH_N 4096
#define NCLS 10
#define ALPHA_C 0.1f
#define OMA_C 0.9f
#define MMARGIN 0.5f

// ---------------------------------------------------------------------------
// Build W = (1-m)I - A^T A + (Bm - Bm^T), convert to f16, and pack directly
// into the WMMA A-fragment (16x32 f16) register layout:
//   lane = (n%16) + 16*((k%32/8)&1),  half = ((k%32)/16)*8 + (k%8)
//   frag base = ((n/16)*16 + k/32) * 32 lanes * 16 halves
// ---------------------------------------------------------------------------
__global__ __launch_bounds__(256) void build_w_pack(const float* __restrict__ A,
                                                    const float* __restrict__ Bm,
                                                    _Float16* __restrict__ Wp)
{
  __shared__ float As[16][17];
  __shared__ float Bs[16][17];
  const int tx = threadIdx.x, ty = threadIdx.y;
  const int nBase = blockIdx.x * 16, kBase = blockIdx.y * 16;
  const int n = nBase + ty, k = kBase + tx;
  float acc = 0.f;
  for (int jc = 0; jc < N_DIM; jc += 16) {
    As[ty][tx] = A[(size_t)(jc + ty) * N_DIM + nBase + tx];
    Bs[ty][tx] = A[(size_t)(jc + ty) * N_DIM + kBase + tx];
    __syncthreads();
#pragma unroll
    for (int jj = 0; jj < 16; ++jj)
      acc += As[jj][ty] * Bs[jj][tx];
    __syncthreads();
  }
  float w = -acc + Bm[(size_t)n * N_DIM + k] - Bm[(size_t)k * N_DIM + n]
          + ((n == k) ? (1.0f - MMARGIN) : 0.f);

  const int nt   = n >> 4;
  const int kt   = k >> 5;
  const int ko   = k & 31;
  const int h    = ((ko >> 4) << 3) + (ko & 7);
  const int lhi  = (ko >> 3) & 1;
  const int lane = (n & 15) + (lhi << 4);
  Wp[((size_t)((nt << 4) + kt) * 32 + lane) * 16 + h] = (_Float16)w;
}

// ---------------------------------------------------------------------------
// biasT[b][n] = sum_d U[n][d] * x[b][d] + bvec[n]   (stored transposed, f32)
// D = 784 = 49 * 16 exactly.
// ---------------------------------------------------------------------------
__global__ __launch_bounds__(256) void build_bias(const float* __restrict__ U,
                                                  const float* __restrict__ x,
                                                  const float* __restrict__ bvec,
                                                  float* __restrict__ biasT)
{
  __shared__ float xs[16][17];
  __shared__ float us[16][17];
  const int tx = threadIdx.x, ty = threadIdx.y;
  const int bBase = blockIdx.x * 16, nBase = blockIdx.y * 16;
  float acc = 0.f;
  for (int dc = 0; dc < D_DIM; dc += 16) {
    xs[ty][tx] = x[(size_t)(bBase + ty) * D_DIM + dc + tx];
    us[ty][tx] = U[(size_t)(nBase + ty) * D_DIM + dc + tx];
    __syncthreads();
#pragma unroll
    for (int dd = 0; dd < 16; ++dd)
      acc += xs[ty][dd] * us[tx][dd];
    __syncthreads();
  }
  biasT[(size_t)(bBase + ty) * N_DIM + nBase + tx] = acc + bvec[nBase + tx];
}

// ---------------------------------------------------------------------------
// z0 = 0
// ---------------------------------------------------------------------------
__global__ __launch_bounds__(256) void zero_half(_Float16* __restrict__ p, int ngroups)
{
  int i = blockIdx.x * blockDim.x + threadIdx.x;
  if (i < ngroups) {
    v8h zv = {};
    ((v8h*)p)[i] = zv;
  }
}

// ---------------------------------------------------------------------------
// One forward-backward iteration:
//   zT_out = ReLU(0.9*zT_in + 0.1*(W @ z + bias))   (all transposed [b][n])
// 1024 waves, each computing a 32(n) x 64(b) tile via 16x16x32 f16 WMMA.
// ---------------------------------------------------------------------------
__global__ __launch_bounds__(256) void mon_iter(const _Float16* __restrict__ Wp,
                                                const _Float16* __restrict__ zin,
                                                const float* __restrict__ biasT,
                                                _Float16* __restrict__ zout)
{
  const int lane  = threadIdx.x & 31;
  const int wave  = (blockIdx.x << 3) + (threadIdx.x >> 5);   // 0..1023
  const int ntile = wave & 15;                                // 16 tiles of 32 n
  const int btile = wave >> 4;                                // 64 tiles of 64 b
  const int n0 = ntile << 5;
  const int b0 = btile << 6;
  const int lmod = lane & 15;
  const int lhi  = lane >> 4;

  v8f acc[2][4];
#pragma unroll
  for (int i = 0; i < 2; ++i)
#pragma unroll
    for (int j = 0; j < 4; ++j) {
      v8f zv = {};
      acc[i][j] = zv;
    }

  for (int kt = 0; kt < 16; ++kt) {
    v16h afrag[2], bfrag[4];
#pragma unroll
    for (int i = 0; i < 2; ++i) {
      const int nt = (n0 >> 4) + i;
      afrag[i] = *(const v16h*)(Wp + ((size_t)((nt << 4) + kt) * 32 + lane) * 16);
    }
#pragma unroll
    for (int j = 0; j < 4; ++j) {
      const int bb = b0 + (j << 4) + lmod;
      const int kk = (kt << 5) + (lhi << 3);
      v8h lo = *(const v8h*)(zin + (size_t)bb * N_DIM + kk);
      v8h hi = *(const v8h*)(zin + (size_t)bb * N_DIM + kk + 16);
      bfrag[j] = __builtin_shufflevector(lo, hi, 0,1,2,3,4,5,6,7,8,9,10,11,12,13,14,15);
    }
#pragma unroll
    for (int i = 0; i < 2; ++i)
#pragma unroll
      for (int j = 0; j < 4; ++j)
        acc[i][j] = __builtin_amdgcn_wmma_f32_16x16x32_f16(
            false, afrag[i], false, bfrag[j], (short)0, acc[i][j], false, false);
  }

#pragma unroll
  for (int i = 0; i < 2; ++i)
#pragma unroll
    for (int j = 0; j < 4; ++j) {
      const int bb = b0 + (j << 4) + lmod;
      const int nn = n0 + (i << 4) + (lhi << 3);
      const size_t base = (size_t)bb * N_DIM + nn;
      v8h zo = *(const v8h*)(zin + base);
      v8f bi = *(const v8f*)(biasT + base);
      v8h outv;
#pragma unroll
      for (int e = 0; e < 8; ++e) {
        float r = OMA_C * (float)zo[e] + ALPHA_C * (acc[i][j][e] + bi[e]);
        outv[e] = (_Float16)(r > 0.f ? r : 0.f);
      }
      *(v8h*)(zout + base) = outv;
    }
}

// ---------------------------------------------------------------------------
// logits[b][c] = sum_n zT[b][n] * Wc[c][n] + bc[c]
// ---------------------------------------------------------------------------
__global__ __launch_bounds__(256) void head_kernel(const _Float16* __restrict__ zT,
                                                   const float* __restrict__ Wc,
                                                   const float* __restrict__ bc,
                                                   float* __restrict__ out)
{
  const int b = blockIdx.x * blockDim.x + threadIdx.x;   // 0..4095
  float acc[NCLS];
#pragma unroll
  for (int c = 0; c < NCLS; ++c) acc[c] = 0.f;
  const _Float16* zr = zT + (size_t)b * N_DIM;
  for (int n = 0; n < N_DIM; ++n) {
    float zv = (float)zr[n];
#pragma unroll
    for (int c = 0; c < NCLS; ++c) acc[c] += zv * Wc[c * N_DIM + n];
  }
#pragma unroll
  for (int c = 0; c < NCLS; ++c) out[b * NCLS + c] = acc[c] + bc[c];
}

// ---------------------------------------------------------------------------
extern "C" void kernel_launch(void* const* d_in, const int* in_sizes, int n_in,
                              void* d_out, int out_size, void* d_ws, size_t ws_size,
                              hipStream_t stream)
{
  const float* x  = (const float*)d_in[0];   // (4096, 784)
  const float* A  = (const float*)d_in[1];   // (512, 512)
  const float* Bm = (const float*)d_in[2];   // (512, 512)
  const float* U  = (const float*)d_in[3];   // (512, 784)
  const float* bv = (const float*)d_in[4];   // (512, 1)
  const float* Wc = (const float*)d_in[5];   // (10, 512)
  const float* bc = (const float*)d_in[6];   // (10,)
  float* out = (float*)d_out;                // (4096, 10)

  char* ws = (char*)d_ws;
  const size_t WP_BYTES   = (size_t)N_DIM * N_DIM * 2;                 // 512 KB
  const size_t BIAS_BYTES = (size_t)BATCH_N * N_DIM * 4;               // 8 MB
  const size_t Z_BYTES    = (size_t)BATCH_N * N_DIM * 2;               // 4 MB
  _Float16* Wp    = (_Float16*)(ws);
  float*    biasT = (float*)(ws + WP_BYTES);
  _Float16* z0    = (_Float16*)(ws + WP_BYTES + BIAS_BYTES);
  _Float16* z1    = (_Float16*)(ws + WP_BYTES + BIAS_BYTES + Z_BYTES);

  dim3 blk16(16, 16);
  build_w_pack<<<dim3(N_DIM / 16, N_DIM / 16), blk16, 0, stream>>>(A, Bm, Wp);
  build_bias<<<dim3(BATCH_N / 16, N_DIM / 16), blk16, 0, stream>>>(U, x, bv, biasT);

  const int ngroups = BATCH_N * N_DIM / 8;   // 262144
  zero_half<<<(ngroups + 255) / 256, 256, 0, stream>>>(z0, ngroups);

  _Float16* zin = z0;
  _Float16* zout = z1;
  for (int it = 0; it < 100; ++it) {
    mon_iter<<<128, 256, 0, stream>>>(Wp, zin, biasT, zout);
    _Float16* t = zin; zin = zout; zout = t;
  }

  head_kernel<<<BATCH_N / 256, 256, 0, stream>>>(zin, Wc, bc, out);
}
